// TFMambaLite_18562848653365
// MI455X (gfx1250) — compile-verified, matrix operand order checked by reading, not run on hardware
//
#include <hip/hip_runtime.h>
#include <hip/hip_bf16.h>

typedef __attribute__((ext_vector_type(16))) _Float16 v16h;
typedef __attribute__((ext_vector_type(8)))  float    v8f;
typedef __attribute__((ext_vector_type(2)))  float    v2f;

#if __has_builtin(__builtin_amdgcn_wmma_f32_16x16x4_f32)
#define HAVE_WMMA_F32X4 1
#endif

#define HIMG 2596
#define WIMG 1286

// ---------------------------------------------------------------------------
// Patch-embed implicit-im2col GEMM, f16 WMMA (16x16x32), f32 accumulate.
// MODE 0: t-patch  k=(h,kw): x[b, h, 10*l+kw],  K = 2596*16
// MODE 1: f-patch  k=(hh,w): x[b, 20*l+hh, w],  K = 36*1286
// One wave = one 16(l) x 16(e) tile of out[b*L+l, e].
// ---------------------------------------------------------------------------
template <int MODE>
__global__ void patch_gemm(const float* __restrict__ x,
                           const float* __restrict__ w,
                           const float* __restrict__ bias,
                           float* __restrict__ out,
                           int L, int K, int numTiles) {
  int wave = blockIdx.x * (blockDim.x >> 5) + (threadIdx.x >> 5);
  if (wave >= numTiles) return;                 // uniform per wave: EXEC stays all-1s
  int lane = threadIdx.x & 31;
  int LT = (L + 15) >> 4;
  int b = wave / (LT * 16);
  int rem = wave - b * LT * 16;
  int l0 = (rem >> 4) << 4;
  int e0 = (rem & 15) << 4;

  int mrow = lane & 15;
  int kb = (lane < 16) ? 0 : 8;                 // K-half split per ISA A/B layout
  int l = l0 + mrow;
  int lc = min(l, L - 1);
  bool rowok = (l < L);
  int e = e0 + (lane & 15);
  const float* wrow = w + (size_t)e * K;

  v8f acc = {0.f, 0.f, 0.f, 0.f, 0.f, 0.f, 0.f, 0.f};
  int Kp = (K + 31) & ~31;
  for (int k0 = 0; k0 < Kp; k0 += 32) {
    v16h af, bf;
#pragma unroll
    for (int j = 0; j < 16; ++j) {
      int k = k0 + kb + j + ((j >= 8) ? 8 : 0); // halves 0..7 -> kb+0..7, 8..15 -> kb+16..23
      bool ok = (k < K);
      int kc = ok ? k : (K - 1);
      size_t xi;
      if (MODE == 0) {
        int h = kc >> 4, kw = kc & 15;
        xi = ((size_t)b * HIMG + h) * WIMG + (10 * lc + kw);
      } else {
        int hh = kc / WIMG;
        int ww = kc - hh * WIMG;
        xi = ((size_t)b * HIMG + 20 * lc + hh) * WIMG + ww;
      }
      float av = (ok && rowok) ? x[xi] : 0.0f;
      float bv = ok ? wrow[kc] : 0.0f;
      af[j] = (_Float16)av;
      bf[j] = (_Float16)bv;
    }
    if (k0 + 32 < Kp) __builtin_prefetch(wrow + k0 + 32 + kb, 0, 3); // global_prefetch_b8
    acc = __builtin_amdgcn_wmma_f32_16x16x32_f16(false, af, false, bf,
                                                 (short)0, acc, false, false);
  }
  float bi = bias[e];
#pragma unroll
  for (int r = 0; r < 8; ++r) {
    int mm = r + ((lane < 16) ? 0 : 8);
    int ll = l0 + mm;
    if (ll < L) out[((size_t)b * L + ll) * 256 + e] = acc[r] + bi;
  }
}

// ---------------------------------------------------------------------------
// Generic token GEMM: out(MxN) = A(MxK, row stride lda) @ W(NxK)^T
//                                [+ bias[n]] [+ Cres(MxN)] [relu]
// Exact f32 via v_wmma_f32_16x16x4_f32 when available; f16 WMMA fallback.
// M, N multiples of 16; K multiple of 4 at every call site.
// ---------------------------------------------------------------------------
__global__ void gemm_tok(const float* __restrict__ A, int lda,
                         const float* __restrict__ W,
                         const float* __restrict__ bias,
                         const float* __restrict__ Cres,
                         float* __restrict__ out,
                         int M, int N, int K, int act, int numTiles) {
  int wave = blockIdx.x * (blockDim.x >> 5) + (threadIdx.x >> 5);
  if (wave >= numTiles) return;
  int lane = threadIdx.x & 31;
  int nt = N >> 4;
  int m0 = (wave / nt) << 4;
  int n0 = (wave % nt) << 4;
  int m = m0 + (lane & 15);
  int n = n0 + (lane & 15);

  v8f acc;
#pragma unroll
  for (int r = 0; r < 8; ++r) {
    int mm = m0 + r + ((lane < 16) ? 0 : 8);
    float c0 = Cres ? Cres[(size_t)mm * N + n] : 0.0f;
    if (bias) c0 += bias[n];
    acc[r] = c0;
  }
  const float* arow = A + (size_t)m * lda;
  const float* wrow = W + (size_t)n * K;

#ifdef HAVE_WMMA_F32X4
  int koff = (lane < 16) ? 0 : 2;               // f32 A 16x4: lanes 0-15 K0,1; 16-31 K2,3
  for (int k0 = 0; k0 < K; k0 += 4) {
    v2f av, bv;
    av.x = arow[k0 + koff];
    av.y = arow[k0 + koff + 1];
    bv.x = wrow[k0 + koff];
    bv.y = wrow[k0 + koff + 1];
    acc = __builtin_amdgcn_wmma_f32_16x16x4_f32(false, av, false, bv,
                                                (short)0, acc, false, false);
  }
#else
  int kb = (lane < 16) ? 0 : 8;
  int Kp = (K + 31) & ~31;
  for (int k0 = 0; k0 < Kp; k0 += 32) {
    v16h af, bf;
#pragma unroll
    for (int j = 0; j < 16; ++j) {
      int k = k0 + kb + j + ((j >= 8) ? 8 : 0);
      bool ok = (k < K);
      int kc = ok ? k : (K - 1);
      af[j] = (_Float16)(ok ? arow[kc] : 0.0f);
      bf[j] = (_Float16)(ok ? wrow[kc] : 0.0f);
    }
    acc = __builtin_amdgcn_wmma_f32_16x16x32_f16(false, af, false, bf,
                                                 (short)0, acc, false, false);
  }
#endif
#pragma unroll
  for (int r = 0; r < 8; ++r) {
    int mm = m0 + r + ((lane < 16) ? 0 : 8);
    float v = acc[r];
    if (act) v = fmaxf(v, 0.0f);
    out[(size_t)mm * N + n] = v;
  }
}

// ---------------------------------------------------------------------------
// Elementwise / reduction kernels
// ---------------------------------------------------------------------------
__global__ void rmsnorm_k(const float* __restrict__ x, const float* __restrict__ w,
                          float* __restrict__ out) {
  __shared__ float red[256];
  int t = blockIdx.x, e = threadIdx.x;
  float v = x[(size_t)t * 256 + e];
  red[e] = v * v;
  __syncthreads();
  for (int s = 128; s > 0; s >>= 1) {
    if (e < s) red[e] += red[e + s];
    __syncthreads();
  }
  float scale = rsqrtf(red[0] * (1.0f / 256.0f) + 1e-5f);
  out[(size_t)t * 256 + e] = v * scale * w[e];
}

// depthwise conv (D_CONV=4, causal pad 3) + bias + silu; input = xz[:, :512]
__global__ void conv_silu_k(const float* __restrict__ xz, const float* __restrict__ cw,
                            const float* __restrict__ cb, float* __restrict__ xc, int L) {
  int tid = blockIdx.x * blockDim.x + threadIdx.x;
  int total = 16 * L * 512;
  if (tid >= total) return;
  int token = tid >> 9;
  int d = tid & 511;
  int l = token % L;
  float s = cb[d];
#pragma unroll
  for (int i = 0; i < 4; ++i) {
    int ll = l + i - 3;
    if (ll >= 0) s += cw[d * 4 + i] * xz[(size_t)(token + i - 3) * 1024 + d];
  }
  xc[(size_t)token * 512 + d] = s / (1.0f + __expf(-s));
}

// selective-scan: one lane per (b, d) with 16 states in registers; fuses
// softplus(dt), dA=exp(delta*A), y = sum(state*C) + x*D, and silu(gate).
__global__ void scan_k(const float* __restrict__ dtr, const float* __restrict__ xc,
                       const float* __restrict__ xp, const float* __restrict__ xz,
                       const float* __restrict__ alog, const float* __restrict__ Dw,
                       float* __restrict__ y, int L) {
  int tid = blockIdx.x * blockDim.x + threadIdx.x;
  if (tid >= 16 * 512) return;
  int b = tid >> 9, d = tid & 511;
  float Ar[16], st[16];
#pragma unroll
  for (int n = 0; n < 16; ++n) { Ar[n] = -__expf(alog[d * 16 + n]); st[n] = 0.f; }
  float Dd = Dw[d];
  for (int l = 0; l < L; ++l) {
    size_t tok = (size_t)b * L + l;
    float dv = dtr[tok * 512 + d];
    float delta = (dv > 20.f) ? dv : log1pf(__expf(dv));
    float xv = xc[tok * 512 + d];
    float g = xz[tok * 1024 + 512 + d];
    float dx = delta * xv;
    float accy = 0.f;
#pragma unroll
    for (int n = 0; n < 16; ++n) {
      float Bn = xp[tok * 48 + 16 + n];
      float Cn = xp[tok * 48 + 32 + n];
      float da = __expf(delta * Ar[n]);
      st[n] = da * st[n] + dx * Bn;
      accy += st[n] * Cn;
    }
    float yv = accy + xv * Dd;
    y[tok * 512 + d] = yv * (g / (1.0f + __expf(-g)));
  }
}

// cross-attention: one block per (b, head); K/V staged into LDS with CDNA5
// async memory->LDS b128 transfers (ASYNCcnt), then 2-pass softmax.
__global__ void attn_k(const float* __restrict__ q, const float* __restrict__ kv,
                       float* __restrict__ ao) {
  __shared__ __align__(16) float ks[128 * 64];
  __shared__ __align__(16) float vs[128 * 64];
  int b = blockIdx.x >> 2, h = blockIdx.x & 3;
  int tid = threadIdx.x;
  // Each thread moves 4 contiguous floats (b128) per async op; rows are 64
  // floats so a 4-float chunk never crosses a K/V row.
  for (int idx = tid * 4; idx < 128 * 64; idx += 128 * 4) {
    int j = idx >> 6, i = idx & 63;
    size_t tok = (size_t)b * 128 + j;
    const float* gk = kv + tok * 512 + h * 64 + i;
    const float* gv = kv + tok * 512 + 256 + h * 64 + i;
    // Low 32 bits of the generic LDS pointer are the LDS byte offset
    // (flat-aperture rule).
    unsigned lk = (unsigned)(uintptr_t)(&ks[idx]);
    unsigned lv = (unsigned)(uintptr_t)(&vs[idx]);
    asm volatile("global_load_async_to_lds_b128 %0, %1, off"
                 :: "v"(lk), "v"(gk) : "memory");
    asm volatile("global_load_async_to_lds_b128 %0, %1, off"
                 :: "v"(lv), "v"(gv) : "memory");
  }
#if __has_builtin(__builtin_amdgcn_s_wait_asynccnt)
  __builtin_amdgcn_s_wait_asynccnt(0);
#else
  asm volatile("s_wait_asynccnt 0x0" ::: "memory");
#endif
  __syncthreads();

  size_t tok = (size_t)b * 128 + tid;
  float qr[64];
#pragma unroll
  for (int i = 0; i < 64; ++i) qr[i] = q[tok * 256 + h * 64 + i];
  float mx = -1e30f;
  for (int j = 0; j < 128; ++j) {
    float s = 0.f;
#pragma unroll
    for (int i = 0; i < 64; ++i) s += qr[i] * ks[j * 64 + i];
    mx = fmaxf(mx, s * 0.125f);
  }
  float acc[64];
#pragma unroll
  for (int i = 0; i < 64; ++i) acc[i] = 0.f;
  float den = 0.f;
  for (int j = 0; j < 128; ++j) {
    float s = 0.f;
#pragma unroll
    for (int i = 0; i < 64; ++i) s += qr[i] * ks[j * 64 + i];
    float p = __expf(s * 0.125f - mx);
    den += p;
#pragma unroll
    for (int i = 0; i < 64; ++i) acc[i] += p * vs[j * 64 + i];
  }
  float inv = 1.0f / den;
#pragma unroll
  for (int i = 0; i < 64; ++i) ao[tok * 256 + h * 64 + i] = acc[i] * inv;
}

__global__ void repack_k(const float* __restrict__ fn, float* __restrict__ fn128) {
  int tid = blockIdx.x * blockDim.x + threadIdx.x;
  if (tid >= 2048 * 256) return;
  int b = tid >> 15;
  int rest = tid & 32767;
  fn128[tid] = fn[(size_t)b * (129 * 256) + rest];
}

__global__ void pool_k(const float* __restrict__ po, float* __restrict__ pooled) {
  int b = blockIdx.x, e = threadIdx.x;
  float s = 0.f;
  for (int l = 0; l < 128; ++l) s += po[((size_t)b * 128 + l) * 256 + e];
  pooled[b * 256 + e] = s * (1.0f / 128.0f);
}

__global__ void head2_k(const float* __restrict__ z, const float* __restrict__ w2,
                        const float* __restrict__ b2, float* __restrict__ out) {
  int b = threadIdx.x;
  if (b >= 16) return;
  float s = b2[0];
  for (int c = 0; c < 512; ++c) s += z[b * 512 + c] * w2[c];
  out[b] = s;
}

// ---------------------------------------------------------------------------
extern "C" void kernel_launch(void* const* d_in, const int* in_sizes, int n_in,
                              void* d_out, int out_size, void* d_ws, size_t ws_size,
                              hipStream_t stream) {
  (void)in_sizes; (void)n_in; (void)out_size; (void)ws_size;
  const float* X = (const float*)d_in[0];
  auto in = [&](int i) { return (const float*)d_in[i]; };

  float* ws = (float*)d_ws;
  size_t off = 0;
  auto alloc = [&](size_t n) { float* p = ws + off; off += n; return p; };
  float* TB    = alloc(2048 * 256);   // t tokens
  float* FB    = alloc(2064 * 256);   // f tokens (L=129)
  float* XN    = alloc(2064 * 256);
  float* XZ    = alloc(2064 * 1024);
  float* XC    = alloc(2064 * 512);
  float* XP    = alloc(2064 * 48);
  float* DTb   = alloc(2064 * 512);
  float* Yb    = alloc(2064 * 512);
  float* TN    = alloc(2048 * 256);
  float* FN    = alloc(2064 * 256);
  float* FN128 = alloc(2048 * 256);
  float* Q     = alloc(2048 * 256);
  float* KV    = alloc(2048 * 512);
  float* AO    = alloc(2048 * 256);
  float* PO    = alloc(2048 * 256);
  float* PL    = alloc(16 * 256);
  float* Z     = alloc(16 * 512);

  auto gemm = [&](const float* A, int lda, const float* W, const float* bias,
                  const float* Cres, float* out_, int M, int N, int K, int act) {
    int tiles = (M >> 4) * (N >> 4);
    gemm_tok<<<dim3((tiles + 3) / 4), dim3(128), 0, stream>>>(
        A, lda, W, bias, Cres, out_, M, N, K, act, tiles);
  };

  // patch embeds (heavy GEMMs, f16 WMMA)
  {
    int tiles = 16 * 8 * 16;
    patch_gemm<0><<<dim3((tiles + 3) / 4), dim3(128), 0, stream>>>(
        X, in(1), in(2), TB, 128, 2596 * 16, tiles);
  }
  {
    int tiles = 16 * 9 * 16;
    patch_gemm<1><<<dim3((tiles + 3) / 4), dim3(128), 0, stream>>>(
        X, in(3), in(4), FB, 129, 36 * 1286, tiles);
  }

  // mamba stacks: t-blocks at params idx 5.., f-blocks at 45..
  for (int s = 0; s < 2; ++s) {
    float* buf = s ? FB : TB;
    int L = s ? 129 : 128;
    int T = 16 * L;
    int pb = s ? 45 : 5;
    for (int blk = 0; blk < 4; ++blk) {
      int p = pb + 10 * blk;
      rmsnorm_k<<<dim3(T), dim3(256), 0, stream>>>(buf, in(p + 0), XN);
      gemm(XN, 256, in(p + 1), nullptr, nullptr, XZ, T, 1024, 256, 0);   // in_proj
      {
        int total = T * 512;
        conv_silu_k<<<dim3((total + 255) / 256), dim3(256), 0, stream>>>(
            XZ, in(p + 2), in(p + 3), XC, L);
      }
      gemm(XC, 512, in(p + 4), nullptr, nullptr, XP, T, 48, 512, 0);     // x_proj
      gemm(XP, 48, in(p + 5), in(p + 6), nullptr, DTb, T, 512, 16, 0);   // dt_proj
      scan_k<<<dim3(32), dim3(256), 0, stream>>>(DTb, XC, XP, XZ,
                                                 in(p + 7), in(p + 8), Yb, L);
      gemm(Yb, 512, in(p + 9), nullptr, buf, buf, T, 256, 512, 0);       // out_proj + res
    }
  }

  // cross attention (cross params at 85..91)
  rmsnorm_k<<<dim3(2048), dim3(256), 0, stream>>>(TB, in(85), TN);
  rmsnorm_k<<<dim3(2064), dim3(256), 0, stream>>>(FB, in(85), FN);
  repack_k<<<dim3(2048), dim3(256), 0, stream>>>(FN, FN128);
  gemm(TN, 256, in(86), in(87), nullptr, Q, 2048, 256, 256, 0);
  gemm(FN128, 256, in(88), in(89), nullptr, KV, 2048, 512, 256, 0);
  attn_k<<<dim3(64), dim3(128), 0, stream>>>(Q, KV, AO);
  gemm(AO, 256, in(90), in(91), TN, PO, 2048, 256, 256, 0);              // proj + tn

  // pool + head (92..95)
  pool_k<<<dim3(16), dim3(256), 0, stream>>>(PO, PL);
  gemm(PL, 256, in(92), in(93), nullptr, Z, 16, 512, 256, 1);            // relu MLP
  head2_k<<<dim3(1), dim3(32), 0, stream>>>(Z, in(94), in(95), (float*)d_out);
}
